// RoPEAttention_49984829391423
// MI455X (gfx1250) — compile-verified, matrix operand order checked by reading, not run on hardware
//
#include <hip/hip_runtime.h>
#include <hip/hip_bf16.h>

typedef __bf16 bf16_t;
typedef __attribute__((ext_vector_type(16))) __bf16 v16bf;
typedef __attribute__((ext_vector_type(8)))  float  v8f;

typedef unsigned int u32;
typedef unsigned long long u64;
typedef __attribute__((ext_vector_type(4))) u32 v4u32;
typedef __attribute__((ext_vector_type(8))) int v8i32;
typedef __attribute__((ext_vector_type(4))) int v4i32;

union B16Frag { v16bf v; uint4 u[2]; };

__device__ __forceinline__ v8f v8f_zero() {
  v8f z;
#pragma unroll
  for (int i = 0; i < 8; ++i) z[i] = 0.0f;
  return z;
}

#define WMMA_BF16(A, B, C) \
  __builtin_amdgcn_wmma_f32_16x16x32_bf16(false, (A), false, (B), (short)0, (C), false, false)

// --------------------------------------------------------------- TDM helper
// 2D tile load Global->LDS via Tensor Data Mover (bf16 elements).
// D# built per CDNA5 ISA 8.3/8.4: group0 {count=1, lds_addr, global_addr, type=2},
// group1 {data_size=1(2B), tensor_dim0/1, tile_dim0/1, tensor_dim0_stride}.
// 6-arg builtin form (clang-23 / therock-10.0 headers).
__device__ __forceinline__ void tdm_load_2d_bf16(
    u32 lds_addr, const void* gptr,
    u32 tensor_d0, u32 tensor_d1, u32 tile_d0, u32 tile_d1, u64 stride0) {
  const u64 ga = (u64)(uintptr_t)gptr;
  v4u32 g0;
  g0[0] = 1u;                                   // count=1, user mode
  g0[1] = lds_addr;                             // LDS byte address
  g0[2] = (u32)ga;                              // global_addr[31:0]
  g0[3] = (u32)(ga >> 32) | 0x80000000u;        // global_addr[56:32] | type=2

  v8i32 g1;
  g1[0] = (int)(1u << 16);                                    // data_size=1 (2B)
  g1[1] = (int)((tensor_d0 & 0xFFFFu) << 16);                 // tensor_dim0[15:0]
  g1[2] = (int)((tensor_d0 >> 16) | ((tensor_d1 & 0xFFFFu) << 16));
  g1[3] = (int)((tensor_d1 >> 16) | (tile_d0 << 16));         // tile_dim0
  g1[4] = (int)(tile_d1 & 0xFFFFu);                           // tile_dim1, tile_dim2=0
  g1[5] = (int)(u32)stride0;                                  // dim0_stride[31:0]
  g1[6] = (int)(u32)(stride0 >> 32);                          // dim0_stride[47:32]
  g1[7] = 0;

  v4i32 z4 = {0, 0, 0, 0};
  v8i32 z8 = {0, 0, 0, 0, 0, 0, 0, 0};
  __builtin_amdgcn_tensor_load_to_lds(g0, g1, z4, z4, z8, 0);
}

// ---------------------------------------------------------------- fp32->bf16
__global__ void cvt_f32_bf16(const float* __restrict__ in, bf16_t* __restrict__ out, int n) {
  int i = blockIdx.x * blockDim.x + threadIdx.x;
  int stride = gridDim.x * blockDim.x;
  for (; i < n; i += stride) out[i] = (bf16_t)in[i];
}

// ------------------------------------------------- QKV GEMM + fused RoPE
// grid: (6272/128, 3072/64) = (49, 48), block 256 (8 waves).
__global__ __launch_bounds__(256) void qkv_rope(
    const bf16_t* __restrict__ xb, const bf16_t* __restrict__ wqb,
    bf16_t* __restrict__ qbuf, bf16_t* __restrict__ kbuf, bf16_t* __restrict__ vtbuf) {
  const int lane = threadIdx.x & 31;
  const int wave = threadIdx.x >> 5;
  const int l15  = lane & 15;
  const int half = lane >> 4;
  const int kb   = half << 3;

  const int m0 = blockIdx.x * 128 + wave * 16;
  const int o0 = blockIdx.y * 64;

  v8f acc[4];
#pragma unroll
  for (int t = 0; t < 4; ++t) acc[t] = v8f_zero();

  const bf16_t* arow  = xb  + (size_t)(m0 + l15) * 1024 + kb;
  const bf16_t* brow0 = wqb + (size_t)(o0 + l15) * 1024 + kb;

  for (int k0 = 0; k0 < 1024; k0 += 32) {
    __builtin_prefetch(arow + k0 + 128, 0, 0);
    B16Frag a;
    a.u[0] = *(const uint4*)(arow + k0);
    a.u[1] = *(const uint4*)(arow + k0 + 16);
#pragma unroll
    for (int t = 0; t < 4; ++t) {
      const bf16_t* br = brow0 + (size_t)t * 16 * 1024 + k0;
      B16Frag b;
      b.u[0] = *(const uint4*)(br);
      b.u[1] = *(const uint4*)(br + 16);
      acc[t] = WMMA_BF16(a.v, b.v, acc[t]);
    }
  }

  const int bidx   = m0 / 1568;
  const int n_base = (m0 % 1568) + half * 8;
  const int which  = o0 >> 10;
  const int head   = (o0 >> 6) & 15;

  if (which == 2) {
#pragma unroll
    for (int t = 0; t < 4; ++t) {
      const int d = t * 16 + l15;
      union { bf16_t h[8]; uint4 u; } tmp;
#pragma unroll
      for (int r = 0; r < 8; ++r) tmp.h[r] = (bf16_t)acc[t][r];
      *(uint4*)(vtbuf + ((size_t)(bidx * 16 + head) * 64 + d) * 1568 + n_base) = tmp.u;
    }
  } else {
    bf16_t* dst = (which == 0) ? qbuf : kbuf;
#pragma unroll
    for (int t = 0; t < 4; ++t) {
      const int d    = t * 16 + l15;
      const bool rot = d < 60;
      const int axis = d / 20;
      const int dd   = d - axis * 20;
      const int jj   = dd % 10;
      const float omega = __expf(-0.9210340371976184f * (float)jj);
      const float sgn   = (d & 1) ? 1.0f : -1.0f;
      bf16_t* drow = dst + ((size_t)(bidx * 16 + head) * 1568) * 64 + d;
#pragma unroll
      for (int r = 0; r < 8; ++r) {
        const float val = acc[t][r];
        const float pv  = __shfl_xor(val, 1, 32);
        const int   n   = n_base + r;
        float o = val;
        if (rot) {
          const int infr = n % 196;
          const float pos = (axis == 0) ? (float)(n / 196)
                           : (axis == 1) ? (float)(infr / 14)
                                         : (float)(infr % 14);
          float s, c;
          __sincosf(pos * omega, &s, &c);
          o = val * c + sgn * pv * s;
        }
        drow[(size_t)n * 64] = (bf16_t)o;
      }
    }
  }
}

// ------------------------------------------------- flash attention (TDM-staged)
// grid: (13, 16 heads, 4 batch), block 256 (8 waves, 16 q-rows each)
__global__ __launch_bounds__(256) void attn(
    const bf16_t* __restrict__ qbuf, const bf16_t* __restrict__ kbuf,
    const bf16_t* __restrict__ vtbuf, bf16_t* __restrict__ ao) {
  __shared__ __attribute__((aligned(16))) bf16_t kTile[32][64];   // 4 KB, [key][dim]
  __shared__ __attribute__((aligned(16))) bf16_t vTile[64][32];   // 4 KB, [dim][key]
  __shared__ __attribute__((aligned(16))) bf16_t pStage[8][16][32]; // 8 KB per-wave P
  const int lane = threadIdx.x & 31;
  const int wave = threadIdx.x >> 5;
  const int l15  = lane & 15;
  const int half = lane >> 4;
  const int kb   = half << 3;

  const int b  = blockIdx.z;
  const int h  = blockIdx.y;
  int q0 = blockIdx.x * 128 + wave * 16;
  if (q0 > 1552) q0 = 1552;   // tail block: duplicate work, keep barrier uniformity

  const size_t bh = (size_t)(b * 16 + h);
  const bf16_t* qrow  = qbuf + (bh * 1568 + q0 + l15) * 64;
  const bf16_t* kbase = kbuf + bh * 1568 * 64;
  const bf16_t* vbase = vtbuf + bh * 64 * 1568;

  B16Frag qa0, qa1;
  qa0.u[0] = *(const uint4*)(qrow + kb);
  qa0.u[1] = *(const uint4*)(qrow + kb + 16);
  qa1.u[0] = *(const uint4*)(qrow + 32 + kb);
  qa1.u[1] = *(const uint4*)(qrow + 32 + kb + 16);

  float mrow[8], lrow[8];
#pragma unroll
  for (int r = 0; r < 8; ++r) { mrow[r] = -1e30f; lrow[r] = 0.0f; }
  v8f acc[4];
#pragma unroll
  for (int t = 0; t < 4; ++t) acc[t] = v8f_zero();

  const float scale = 0.125f;
  const u32 kTileLds = (u32)(uintptr_t)&kTile[0][0];
  const u32 vTileLds = (u32)(uintptr_t)&vTile[0][0];

  for (int j0 = 0; j0 < 1568; j0 += 32) {       // 49 exact chunks of 32 keys
    if (wave == 0) {
      // K chunk: 32 rows x 64 dims, row-major stride 64
      tdm_load_2d_bf16(kTileLds, kbase + (size_t)j0 * 64, 64, 1568, 64, 32, 64);
      // V chunk: 64 d-rows x 32 keys from vt[d][n], stride 1568
      tdm_load_2d_bf16(vTileLds, vbase + j0, 1568, 64, 32, 64, 1568);
      __builtin_amdgcn_s_wait_tensorcnt(0);
    }
    __syncthreads();

    v8f s0 = v8f_zero(), s1 = v8f_zero();
    {
      B16Frag b0, b1;
      const bf16_t* kr = &kTile[l15][0];
      b0.u[0] = *(const uint4*)(kr + kb);
      b0.u[1] = *(const uint4*)(kr + kb + 16);
      b1.u[0] = *(const uint4*)(kr + 32 + kb);
      b1.u[1] = *(const uint4*)(kr + 32 + kb + 16);
      s0 = WMMA_BF16(qa0.v, b0.v, s0);
      s0 = WMMA_BF16(qa1.v, b1.v, s0);
      kr = &kTile[16 + l15][0];
      b0.u[0] = *(const uint4*)(kr + kb);
      b0.u[1] = *(const uint4*)(kr + kb + 16);
      b1.u[0] = *(const uint4*)(kr + 32 + kb);
      b1.u[1] = *(const uint4*)(kr + 32 + kb + 16);
      s1 = WMMA_BF16(qa0.v, b0.v, s1);
      s1 = WMMA_BF16(qa1.v, b1.v, s1);
    }

    float alpha[8];
#pragma unroll
    for (int r = 0; r < 8; ++r) {
      float a0 = s0[r] * scale;
      float a1 = s1[r] * scale;
      float mx = fmaxf(a0, a1);
#pragma unroll
      for (int msk = 1; msk <= 8; msk <<= 1)
        mx = fmaxf(mx, __shfl_xor(mx, msk, 32));
      const float mnew = fmaxf(mrow[r], mx);
      alpha[r] = __expf(mrow[r] - mnew);
      mrow[r]  = mnew;
      const float p0 = __expf(a0 - mnew);
      const float p1 = __expf(a1 - mnew);
      float rs = p0 + p1;
#pragma unroll
      for (int msk = 1; msk <= 8; msk <<= 1)
        rs += __shfl_xor(rs, msk, 32);
      lrow[r] = lrow[r] * alpha[r] + rs;
      s0[r] = p0;
      s1[r] = p1;
    }
#pragma unroll
    for (int t = 0; t < 4; ++t)
#pragma unroll
      for (int r = 0; r < 8; ++r) acc[t][r] *= alpha[r];

    // C-layout -> A-layout transpose of P through per-wave LDS region
#pragma unroll
    for (int r = 0; r < 8; ++r) {
      pStage[wave][r + half * 8][l15]      = (bf16_t)s0[r];
      pStage[wave][r + half * 8][16 + l15] = (bf16_t)s1[r];
    }
    __asm__ volatile("s_wait_dscnt 0" ::: "memory");
    B16Frag pa;
    pa.u[0] = *(const uint4*)(&pStage[wave][l15][kb]);
    pa.u[1] = *(const uint4*)(&pStage[wave][l15][kb + 16]);
#pragma unroll
    for (int t = 0; t < 4; ++t) {
      B16Frag vb;
      const bf16_t* vr = &vTile[t * 16 + l15][0];
      vb.u[0] = *(const uint4*)(vr + kb);
      vb.u[1] = *(const uint4*)(vr + kb + 16);
      acc[t] = WMMA_BF16(pa.v, vb.v, acc[t]);
    }
    __syncthreads();   // all waves done with kTile/vTile before next TDM overwrite
  }

#pragma unroll
  for (int t = 0; t < 4; ++t) {
    const int d = t * 16 + l15;
#pragma unroll
    for (int r = 0; r < 8; ++r) {
      const int n = q0 + r + half * 8;
      ao[((size_t)b * 1568 + n) * 1024 + h * 64 + d] = (bf16_t)(acc[t][r] / lrow[r]);
    }
  }
}

// ------------------------------------------------- output projection + bias
__global__ __launch_bounds__(256) void proj(
    const bf16_t* __restrict__ aob, const bf16_t* __restrict__ wpb,
    const float* __restrict__ bias, float* __restrict__ out) {
  const int lane = threadIdx.x & 31;
  const int wave = threadIdx.x >> 5;
  const int l15  = lane & 15;
  const int half = lane >> 4;
  const int kb   = half << 3;

  const int m0 = blockIdx.x * 128 + wave * 16;
  const int o0 = blockIdx.y * 64;

  v8f acc[4];
#pragma unroll
  for (int t = 0; t < 4; ++t) acc[t] = v8f_zero();

  const bf16_t* arow  = aob + (size_t)(m0 + l15) * 1024 + kb;
  const bf16_t* brow0 = wpb + (size_t)(o0 + l15) * 1024 + kb;

  for (int k0 = 0; k0 < 1024; k0 += 32) {
    __builtin_prefetch(arow + k0 + 128, 0, 0);
    B16Frag a;
    a.u[0] = *(const uint4*)(arow + k0);
    a.u[1] = *(const uint4*)(arow + k0 + 16);
#pragma unroll
    for (int t = 0; t < 4; ++t) {
      const bf16_t* br = brow0 + (size_t)t * 16 * 1024 + k0;
      B16Frag b;
      b.u[0] = *(const uint4*)(br);
      b.u[1] = *(const uint4*)(br + 16);
      acc[t] = WMMA_BF16(a.v, b.v, acc[t]);
    }
  }

#pragma unroll
  for (int t = 0; t < 4; ++t) {
    const int o = o0 + t * 16 + l15;
    const float bv = bias[o];
#pragma unroll
    for (int r = 0; r < 8; ++r) {
      const int row = m0 + r + half * 8;
      out[(size_t)row * 1024 + o] = acc[t][r] + bv;
    }
  }
}

// ------------------------------------------------- launcher
extern "C" void kernel_launch(void* const* d_in, const int* in_sizes, int n_in,
                              void* d_out, int out_size, void* d_ws, size_t ws_size,
                              hipStream_t stream) {
  (void)in_sizes; (void)n_in; (void)out_size; (void)ws_size;
  const float* x     = (const float*)d_in[0];
  const float* Wqkv  = (const float*)d_in[1];
  const float* Wproj = (const float*)d_in[2];
  const float* bproj = (const float*)d_in[3];
  float* out = (float*)d_out;

  char* ws = (char*)d_ws;
  size_t off = 0;
  auto alloc = [&](size_t bytes) -> void* {
    void* p = ws + off;
    off = (off + bytes + 255) & ~(size_t)255;
    return p;
  };
  const size_t M = 6272;
  bf16_t* xb  = (bf16_t*)alloc(M * 1024 * 2);
  bf16_t* wqb = (bf16_t*)alloc((size_t)3072 * 1024 * 2);
  bf16_t* wpb = (bf16_t*)alloc((size_t)1024 * 1024 * 2);
  bf16_t* qb  = (bf16_t*)alloc((size_t)4 * 16 * 1568 * 64 * 2);
  bf16_t* kb  = (bf16_t*)alloc((size_t)4 * 16 * 1568 * 64 * 2);
  bf16_t* vtb = (bf16_t*)alloc((size_t)4 * 16 * 64 * 1568 * 2);
  bf16_t* aob = (bf16_t*)alloc(M * 1024 * 2);

  cvt_f32_bf16<<<2048, 256, 0, stream>>>(x,     xb,  (int)(M * 1024));
  cvt_f32_bf16<<<2048, 256, 0, stream>>>(Wqkv,  wqb, 3072 * 1024);
  cvt_f32_bf16<<<1024, 256, 0, stream>>>(Wproj, wpb, 1024 * 1024);

  qkv_rope<<<dim3(49, 48), 256, 0, stream>>>(xb, wqb, qb, kb, vtb);
  attn<<<dim3(13, 16, 4), 256, 0, stream>>>(qb, kb, vtb, aob);
  proj<<<dim3(49, 16), 256, 0, stream>>>(aob, wpb, bproj, out);
}
